// ParallelSelfSEAttention_30331059044904
// MI455X (gfx1250) — matrix operand order repaired and checked by reading, not compiled, not run on hardware
//
#include <hip/hip_runtime.h>
#include <hip/hip_bf16.h>

typedef __attribute__((ext_vector_type(16))) _Float16 v16h;
typedef __attribute__((ext_vector_type(8)))  _Float16 v8h;
typedef __attribute__((ext_vector_type(8)))  float    v8f;

#define BB 8
#define CC 256
#define NN 4096
#define CQ 32
#define CS 16

// ---------------------------------------------------------------------------
// x (B,C,N) f32  ->  xh (B,N,C) f16   (LDS tiled transpose, 64x64 tiles)
// ---------------------------------------------------------------------------
__global__ __launch_bounds__(256) void xpose_kernel(const float* __restrict__ x,
                                                    _Float16* __restrict__ xh) {
  const int b  = blockIdx.z;
  const int c0 = blockIdx.y * 64;
  const int n0 = blockIdx.x * 64;
  __shared__ _Float16 T[64][72];
  const int tid = threadIdx.x;
  const int nl  = tid & 63;
  const int g   = tid >> 6;           // 0..3
  for (int m = 0; m < 16; ++m) {
    const int cl = m * 4 + g;
    T[cl][nl] = (_Float16)x[((size_t)(b * CC + c0 + cl)) * NN + n0 + nl];
  }
  __syncthreads();
  const int cl2 = tid & 63;
  const int g2  = tid >> 6;
  for (int m = 0; m < 16; ++m) {
    const int nl2 = m * 4 + g2;
    xh[((size_t)(b * NN + n0 + nl2)) * CC + c0 + cl2] = T[cl2][nl2];
  }
}

// ---------------------------------------------------------------------------
// Pack Wq(32x256), Wk(32x256), Wv(256x256) -> Wall(320x256) f16, biases f32
// ---------------------------------------------------------------------------
__global__ __launch_bounds__(256) void wconv_kernel(const float* __restrict__ Wq,
                                                    const float* __restrict__ bq,
                                                    const float* __restrict__ Wk,
                                                    const float* __restrict__ bk,
                                                    const float* __restrict__ Wv,
                                                    const float* __restrict__ bv,
                                                    _Float16* __restrict__ Wall,
                                                    float* __restrict__ ball) {
  const int o = blockIdx.x;      // 0..319
  const int c = threadIdx.x;     // 0..255
  float v;
  if (o < 32)       v = Wq[o * CC + c];
  else if (o < 64)  v = Wk[(o - 32) * CC + c];
  else              v = Wv[(o - 64) * CC + c];
  Wall[o * CC + c] = (_Float16)v;
  if (c == 0) ball[o] = (o < 32) ? bq[o] : (o < 64) ? bk[o - 32] : bv[o - 64];
}

// ---------------------------------------------------------------------------
// Per-channel mean of x -> xmean (B*C)
// ---------------------------------------------------------------------------
__global__ __launch_bounds__(256) void mean_kernel(const float* __restrict__ x,
                                                   float* __restrict__ xmean) {
  const int bc = blockIdx.x;
  __shared__ float red[256];
  float s = 0.f;
  for (int i = threadIdx.x; i < NN; i += 256) s += x[(size_t)bc * NN + i];
  red[threadIdx.x] = s;
  __syncthreads();
  for (int st = 128; st > 0; st >>= 1) {
    if (threadIdx.x < st) red[threadIdx.x] += red[threadIdx.x + st];
    __syncthreads();
  }
  if (threadIdx.x == 0) xmean[bc] = red[0] * (1.f / (float)NN);
}

// ---------------------------------------------------------------------------
// SE branch: se[b,c] = sigmoid(W2 @ relu(W1 @ mean))
// ---------------------------------------------------------------------------
__global__ __launch_bounds__(256) void se_kernel(const float* __restrict__ xmean,
                                                 const float* __restrict__ W1,
                                                 const float* __restrict__ W2,
                                                 float* __restrict__ se) {
  const int b = blockIdx.x;
  __shared__ float ym[256];
  __shared__ float y1s[16];
  const int tid = threadIdx.x;
  ym[tid] = xmean[b * CC + tid];
  __syncthreads();
  if (tid < CS) {
    float a = 0.f;
    for (int c = 0; c < CC; ++c) a += W1[tid * CC + c] * ym[c];
    y1s[tid] = a > 0.f ? a : 0.f;
  }
  __syncthreads();
  float a = 0.f;
  for (int j = 0; j < CS; ++j) a += W2[tid * CS + j] * y1s[j];
  se[b * CC + tid] = 1.f / (1.f + __expf(-a));
}

// ---------------------------------------------------------------------------
// Fused QKV: [q;k;v](320,N) = Wall(320,256) @ x(256,N) + bias, per batch.
// ---------------------------------------------------------------------------
__global__ __launch_bounds__(32) void qkv_kernel(const _Float16* __restrict__ Wall,
                                                 const float* __restrict__ ball,
                                                 const _Float16* __restrict__ xh,
                                                 _Float16* __restrict__ qh,
                                                 _Float16* __restrict__ kh,
                                                 _Float16* __restrict__ vh) {
  const int b    = blockIdx.z;
  const int o0   = blockIdx.y * 16;   // 0..304
  const int n0   = blockIdx.x * 16;
  const int lane = threadIdx.x;
  const int half = lane >> 4;
  const int l16  = lane & 15;

  v8f acc;
  for (int e = 0; e < 8; ++e) acc[e] = 0.f;

  for (int k0 = 0; k0 < CC; k0 += 32) {
    v16h a;
    {
      const _Float16* wr = Wall + (size_t)(o0 + l16) * CC + k0;
      v8h lo = *(const v8h*)(wr + half * 16);
      v8h hi = *(const v8h*)(wr + half * 16 + 8);
      for (int e = 0; e < 8; ++e) { a[e] = lo[e]; a[8 + e] = hi[e]; }
    }
    v16h bm;
    {
      const _Float16* xr = xh + ((size_t)(b * NN + n0 + l16)) * CC + k0 + half * 16;
      v8h a0 = *(const v8h*)xr;
      v8h a1 = *(const v8h*)(xr + 8);
      for (int e = 0; e < 8; ++e) { bm[e] = a0[e]; bm[8 + e] = a1[e]; }
    }
    acc = __builtin_amdgcn_wmma_f32_16x16x32_f16(false, a, false, bm,
                                                 (short)0, acc, false, false);
  }

  for (int r = 0; r < 8; ++r) {
    const int o = o0 + r + half * 8;
    const int n = n0 + l16;
    const float v = acc[r] + ball[o];
    const _Float16 hv = (_Float16)v;
    if (o0 < 32)      qh[((size_t)(b * NN + n)) * CQ + o]        = hv;
    else if (o0 < 64) kh[((size_t)(b * NN + n)) * CQ + (o - 32)] = hv;
    else              vh[((size_t)(b * CC + (o - 64))) * NN + n] = hv;
  }
}

// ---------------------------------------------------------------------------
// Flash attention, 4 waves / block, 64 query rows / block.
// K(2KB)/V(16KB) j-tiles staged in LDS via double-buffered async DMA
// (global_load_async_to_lds_b128, ASYNCcnt), shared by all 4 waves.
// ---------------------------------------------------------------------------
__global__ __launch_bounds__(128) void attn_kernel(const _Float16* __restrict__ qh,
                                                   const _Float16* __restrict__ kh,
                                                   const _Float16* __restrict__ vh,
                                                   const float* __restrict__ x,
                                                   const float* __restrict__ se,
                                                   const float* __restrict__ gamma,
                                                   float* __restrict__ out) {
  const int b    = blockIdx.y;
  const int tid  = threadIdx.x;
  const int w    = tid >> 5;          // wave 0..3
  const int lane = tid & 31;
  const int half = lane >> 4;
  const int l16  = lane & 15;
  const int i0   = blockIdx.x * 64 + w * 16;

  __shared__ __align__(16) _Float16 Klds[2][32][32];    // [buf][j][d]
  __shared__ __align__(16) _Float16 Vlds[2][256][32];   // [buf][c][j]
  __shared__ __align__(16) _Float16 Plds[4][16][32];    // per-wave P tile
  __shared__ float Olds[4][16][17];                     // per-wave O transpose

  // Q A-tile (16 rows i x 32 d), reused for every j-tile
  v16h aq;
  {
    const _Float16* qr = qh + ((size_t)(b * NN + i0 + l16)) * CQ;
    v8h lo = *(const v8h*)(qr + half * 16);
    v8h hi = *(const v8h*)(qr + half * 16 + 8);
    for (int e = 0; e < 8; ++e) { aq[e] = lo[e]; aq[8 + e] = hi[e]; }
  }

  v8f acc[16];
  for (int t = 0; t < 16; ++t)
    for (int e = 0; e < 8; ++e) acc[t][e] = 0.f;
  float m[8], l[8];
  for (int r = 0; r < 8; ++r) { m[r] = -1e30f; l[r] = 0.f; }

  v8f zc;
  for (int e = 0; e < 8; ++e) zc[e] = 0.f;

  // async DMA of one 32-j K/V tile into LDS buffer bi (cooperative, 16B/lane)
  auto issue_loads = [&](int bi, int j0) {
    const unsigned vbase = (unsigned)(size_t)&Vlds[bi][0][0];
    for (int i2 = 0; i2 < 8; ++i2) {
      const int chunk = w * 256 + i2 * 32 + lane;      // 0..1023
      const int c    = chunk >> 2;
      const int part = chunk & 3;
      const unsigned long long ga =
          (unsigned long long)(size_t)(vh + ((size_t)(b * CC + c)) * NN + j0 + part * 8);
      const unsigned lo = vbase + chunk * 16;
      asm volatile("global_load_async_to_lds_b128 %0, %1, off"
                   :: "v"(lo), "v"(ga) : "memory");
    }
    const unsigned kbase = (unsigned)(size_t)&Klds[bi][0][0];
    {
      const int chunk = w * 32 + lane;                 // 0..127
      const int j    = chunk >> 2;
      const int part = chunk & 3;
      const unsigned long long ga =
          (unsigned long long)(size_t)(kh + ((size_t)(b * NN + j0 + j)) * CQ + part * 8);
      const unsigned lo = kbase + chunk * 16;
      asm volatile("global_load_async_to_lds_b128 %0, %1, off"
                   :: "v"(lo), "v"(ga) : "memory");
    }
  };

  issue_loads(0, 0);

  for (int jt = 0; jt < 128; ++jt) {
    const int cur = jt & 1;
    const int j0  = jt * 32;

    // buffer `cur` DMA complete (own wave), then block-wide rendezvous
    asm volatile("s_wait_asynccnt 0x0" ::: "memory");
    __syncthreads();

    // overlap: kick off next tile's DMA into the other buffer
    if (jt + 1 < 128) issue_loads(cur ^ 1, j0 + 32);

    // --- energy = Q^T K from LDS K tile ---
    v16h bk0, bk1;
    {
      const _Float16* kr0 = &Klds[cur][l16][half * 16];
      const _Float16* kr1 = &Klds[cur][16 + l16][half * 16];
      v8h a0 = *(const v8h*)kr0, a1 = *(const v8h*)(kr0 + 8);
      v8h b0 = *(const v8h*)kr1, b1 = *(const v8h*)(kr1 + 8);
      for (int e = 0; e < 8; ++e) {
        bk0[e] = a0[e]; bk0[8 + e] = a1[e];
        bk1[e] = b0[e]; bk1[8 + e] = b1[e];
      }
    }
    v8f e0 = __builtin_amdgcn_wmma_f32_16x16x32_f16(false, aq, false, bk0,
                                                    (short)0, zc, false, false);
    v8f e1 = __builtin_amdgcn_wmma_f32_16x16x32_f16(false, aq, false, bk1,
                                                    (short)0, zc, false, false);

    // --- online softmax (row i = r + 8*half, col = lane%16 within group) ---
    for (int r = 0; r < 8; ++r) {
      const float x0 = e0[r], x1 = e1[r];
      float t = fmaxf(x0, x1);
      t = fmaxf(t, __shfl_xor(t, 1, 32));
      t = fmaxf(t, __shfl_xor(t, 2, 32));
      t = fmaxf(t, __shfl_xor(t, 4, 32));
      t = fmaxf(t, __shfl_xor(t, 8, 32));
      const float mn = fmaxf(m[r], t);
      const float sc = __expf(m[r] - mn);
      const float p0 = __expf(x0 - mn);
      const float p1 = __expf(x1 - mn);
      float s = p0 + p1;
      s += __shfl_xor(s, 1, 32);
      s += __shfl_xor(s, 2, 32);
      s += __shfl_xor(s, 4, 32);
      s += __shfl_xor(s, 8, 32);
      l[r] = l[r] * sc + s;
      m[r] = mn;
      for (int t2 = 0; t2 < 16; ++t2) acc[t2][r] *= sc;
      const int il = r + half * 8;
      Plds[w][il][l16]      = (_Float16)p0;
      Plds[w][il][16 + l16] = (_Float16)p1;
    }
    // P tile is per-wave; DS is in-order per wave — just fence the compiler
    // and drain DScnt before re-reading it with the A-operand layout.
    asm volatile("s_wait_dscnt 0x0" ::: "memory");

    v16h ap;
    {
      const _Float16* pr = &Plds[w][l16][0];
      v8h lo = *(const v8h*)(pr + half * 16);
      v8h hi = *(const v8h*)(pr + half * 16 + 8);
      for (int e = 0; e < 8; ++e) { ap[e] = lo[e]; ap[8 + e] = hi[e]; }
    }

    // --- O(16x256) += P(16x32) @ V^T(32x256), B operand from LDS V tile ---
    for (int t = 0; t < 16; ++t) {
      const _Float16* vr = &Vlds[cur][t * 16 + l16][half * 16];
      v16h bv;
      v8h a0 = *(const v8h*)vr;
      v8h a1 = *(const v8h*)(vr + 8);
      for (int e = 0; e < 8; ++e) { bv[e] = a0[e]; bv[8 + e] = a1[e]; }
      acc[t] = __builtin_amdgcn_wmma_f32_16x16x32_f16(false, ap, false, bv,
                                                      (short)0, acc[t],
                                                      false, false);
    }
  }

  // --- epilogue: out = gamma*O/l + x*(1+se), coalesced via per-wave LDS ---
  float inv[8];
  for (int r = 0; r < 8; ++r) inv[r] = 1.f / l[r];
  const float g = gamma[0];

  for (int t = 0; t < 16; ++t) {
    for (int r = 0; r < 8; ++r)
      Olds[w][l16][r + half * 8] = acc[t][r] * inv[r];
    asm volatile("s_wait_dscnt 0x0" ::: "memory");
    for (int pass = 0; pass < 8; ++pass) {
      const int cl = pass * 2 + half;
      const int il = l16;
      const int cg = t * 16 + cl;
      const size_t idx = ((size_t)(b * CC + cg)) * NN + i0 + il;
      const float sev = se[b * CC + cg];
      out[idx] = g * Olds[w][cl][il] + x[idx] * (1.f + sev);
    }
    asm volatile("" ::: "memory");
  }
}

// ---------------------------------------------------------------------------
extern "C" void kernel_launch(void* const* d_in, const int* in_sizes, int n_in,
                              void* d_out, int out_size, void* d_ws, size_t ws_size,
                              hipStream_t stream) {
  const float* x     = (const float*)d_in[0];
  const float* Wq    = (const float*)d_in[1];
  const float* bq    = (const float*)d_in[2];
  const float* Wk    = (const float*)d_in[3];
  const float* bk    = (const float*)d_in[4];
  const float* Wv    = (const float*)d_in[5];
  const float* bv    = (const float*)d_in[6];
  const float* gamma = (const float*)d_in[7];
  const float* W1    = (const float*)d_in[8];
  const float* W2    = (const float*)d_in[9];
  float* out = (float*)d_out;

  char* p = (char*)d_ws;
  auto alloc = [&](size_t bytes) -> void* {
    void* r = (void*)p;
    p += (bytes + 255) & ~(size_t)255;
    return r;
  };
  _Float16* xh    = (_Float16*)alloc((size_t)BB * NN * CC * 2);  // 16 MB
  _Float16* qh    = (_Float16*)alloc((size_t)BB * NN * CQ * 2);  //  2 MB
  _Float16* kh    = (_Float16*)alloc((size_t)BB * NN * CQ * 2);  //  2 MB
  _Float16* vh    = (_Float16*)alloc((size_t)BB * CC * NN * 2);  // 16 MB
  _Float16* Wall  = (_Float16*)alloc((size_t)320 * CC * 2);
  float*    ball  = (float*)alloc(320 * 4);
  float*    xmean = (float*)alloc((size_t)BB * CC * 4);
  float*    sew   = (float*)alloc((size_t)BB * CC * 4);
  (void)ws_size; (void)in_sizes; (void)n_in; (void)out_size;

  xpose_kernel<<<dim3(NN / 64, CC / 64, BB), 256, 0, stream>>>(x, xh);
  wconv_kernel<<<dim3(320), 256, 0, stream>>>(Wq, bq, Wk, bk, Wv, bv, Wall, ball);
  mean_kernel<<<dim3(BB * CC), 256, 0, stream>>>(x, xmean);
  se_kernel<<<dim3(BB), 256, 0, stream>>>(xmean, W1, W2, sew);

  qkv_kernel<<<dim3(NN / 16, 320 / 16, BB), 32, 0, stream>>>(Wall, ball, xh,
                                                             qh, kh, vh);

  attn_kernel<<<dim3(NN / 64, BB), 128, 0, stream>>>(qh, kh, vh, x, sew,
                                                     gamma, out);
}
